// AttentionNet_72353019068556
// MI455X (gfx1250) — compile-verified, matrix operand order checked by reading, not run on hardware
//
#include <hip/hip_runtime.h>
#include <cstdint>

typedef __bf16 bf16;
typedef __attribute__((ext_vector_type(16))) __bf16 v16bf;
typedef __attribute__((ext_vector_type(8)))  float  v8f;

#define S_TOT 829
#define BATCH 32
#define CH    200
#define NH    8
#define HD    32
#define RH    100
#define G4    400

#if defined(__gfx1250__) && __has_builtin(__builtin_amdgcn_tensor_load_to_lds)
#define USE_TDM 1
#else
#define USE_TDM 0
#endif

#if USE_TDM
typedef __attribute__((ext_vector_type(4))) unsigned int ui4;
typedef __attribute__((ext_vector_type(8))) int i8v;
typedef __attribute__((ext_vector_type(4))) int i4v;

// Issue one TDM 2D tile load (bf16 elements) global -> LDS.
//   tile_w x tile_h tile, global row stride row_stride (elements),
//   tensor dims tw x th relative to tile start (hardware zero-fills OOB),
//   pad=1 adds 16B LDS padding per 64B row (-> 40-element LDS row stride).
__device__ inline void tdm_load_2d(unsigned int lds_addr, const void* gptr,
                                   unsigned int tile_w, unsigned int tile_h,
                                   unsigned int row_stride, unsigned int tw,
                                   unsigned int th, int pad)
{
  unsigned long long ga = (unsigned long long)(uintptr_t)gptr;
  ui4 g0 = { 1u,                                   // count=1, user descriptor
             lds_addr,                             // lds_addr [63:32]
             (unsigned int)ga,                     // global_addr lo
             (unsigned int)((ga >> 32) & 0x1ffffffull) | (2u << 30) }; // type=2
  unsigned int dw0 = (1u << 16);                   // data_size = 2 bytes
  if (pad) dw0 |= (1u << 20) | (3u << 22) | (3u << 25); // pad_en, 16DW interval, 4DW pad
  i8v g1 = { (int)dw0,
             (int)(tw << 16),                      // tensor_dim0[15:0] @ bits63:48
             (int)((tw >> 16) | (th << 16)),       // tensor_dim0 hi | tensor_dim1 lo
             (int)((th >> 16) | (tile_w << 16)),   // tensor_dim1 hi | tile_dim0
             (int)(tile_h & 0xffffu),              // tile_dim1 (tile_dim2 = 0)
             (int)row_stride,                      // tensor_dim0_stride[31:0]
             0, 0 };
  i4v z4 = { 0, 0, 0, 0 };
  i8v z8 = { 0, 0, 0, 0, 0, 0, 0, 0 };
  __builtin_amdgcn_tensor_load_to_lds(g0, g1, z4, z4, z8, 0);
}
#endif

// ---------------------------------------------------------------------------
// Conv1d(4->200,k=13) + BN(eval) + ReLU + MaxPool(6) -> bf16 x[s][b][c].
// ---------------------------------------------------------------------------
__global__ __launch_bounds__(256) void conv_branch_kernel(
    const float* __restrict__ in, const float* __restrict__ w,
    const float* __restrict__ gamma, const float* __restrict__ beta,
    const float* __restrict__ mean, const float* __restrict__ var,
    bf16* __restrict__ xout, int Lin, int Spool, int s_base)
{
  int idx = blockIdx.x * 256 + threadIdx.x;
  int total = Spool * BATCH * CH;
  if (idx >= total) return;
  int c = idx % CH;
  int b = (idx / CH) % BATCH;
  int s = idx / (CH * BATCH);
  const float* wp = w + c * 52;
  int t0 = s * 6;
  float scale = gamma[c] * rsqrtf(var[c] + 1e-5f);
  float mu = mean[c], bb = beta[c];
  float best = 0.f;
  for (int j = 0; j < 6; ++j) {
    float acc = 0.f;
    const float* ip = in + (long)b * 4 * Lin + t0 + j;
#pragma unroll
    for (int ic = 0; ic < 4; ++ic)
#pragma unroll
      for (int kk = 0; kk < 13; ++kk)
        acc += ip[ic * Lin + kk] * wp[ic * 13 + kk];
    float y = (acc - mu) * scale + bb;
    best = fmaxf(best, fmaxf(y, 0.f));
  }
  xout[((long)(s_base + s) * BATCH + b) * CH + c] = (bf16)best;
}

// ---------------------------------------------------------------------------
// One-time weight prep: cast/transpose all GEMM B-operands to bf16 (N,K)
// row-major, pad Whh K to 128, sum LSTM biases.
// ---------------------------------------------------------------------------
__global__ __launch_bounds__(256) void prep_kernel(
    const float* __restrict__ wih, const float* __restrict__ whh,
    const float* __restrict__ bih, const float* __restrict__ bhh,
    const float* __restrict__ Qw, const float* __restrict__ Kw,
    const float* __restrict__ Vw, const float* __restrict__ mhw,
    bf16* __restrict__ wihb, bf16* __restrict__ whhb,
    bf16* __restrict__ qwT, bf16* __restrict__ kwT, bf16* __restrict__ vwT,
    bf16* __restrict__ mhwT, float* __restrict__ biassum)
{
  int i = blockIdx.x * 256 + threadIdx.x;
  if (i < 2 * 2 * G4 * CH) wihb[i] = (bf16)wih[i];      // already (N,K)
  if (i < 2 * 2 * G4 * 128) {                           // K padded 100 -> 128
    int k = i & 127, n = (i >> 7) % G4, ld = i / (128 * G4);
    whhb[i] = (bf16)((k < RH) ? whh[((long)ld * G4 + n) * RH + k] : 0.f);
  }
  if (i < NH * HD * CH) {                               // (8,200,32) -> (8,32,200)
    int d = i % CH, e = (i / CH) % HD, h = i / (CH * HD);
    long src = ((long)h * CH + d) * HD + e;
    qwT[i] = (bf16)Qw[src];
    kwT[i] = (bf16)Kw[src];
    vwT[i] = (bf16)Vw[src];
  }
  if (i < RH * NH * HD) {                               // (256,100) -> (100,256)
    int k = i % (NH * HD), n = i / (NH * HD);
    mhwT[i] = (bf16)mhw[(long)k * RH + n];
  }
  if (i < 2 * 2 * G4) biassum[i] = bih[i] + bhh[i];
}

// ---------------------------------------------------------------------------
// Batched bf16 WMMA GEMM:  C = act(alpha*A(MxK)*B^T + bias);  B is (N,K).
// Tiles staged global->LDS by the Tensor Data Mover (edge zero-fill + LDS
// row padding done in hardware). Outputs: f32 C and/or bf16 Cbf (optionally
// transposed). Batch z: zh=z/zdiv, zl=z%zdiv, offset = zh*Hi + zl*Lo.
// ---------------------------------------------------------------------------
__global__ __launch_bounds__(256) void gemm_wmma_kernel(
    const bf16* __restrict__ A, const bf16* __restrict__ Bm,
    const float* __restrict__ bias, float* __restrict__ C,
    bf16* __restrict__ Cbf, int transCbf,
    int M, int N, int K, int lda, int ldb, int ldc, int ldcb,
    float alpha, int act, int zdiv,
    long aHi, long aLo, long bHi, long bLo, long cHi, long cLo,
    long cbHi, long cbLo, long biasHi, long biasLo)
{
  __shared__ __align__(16) bf16 As[64][40];
  __shared__ __align__(16) bf16 Bs[64][40];
  int z = blockIdx.z, zh = z / zdiv, zl = z % zdiv;
  A  += (long)zh * aHi + (long)zl * aLo;
  Bm += (long)zh * bHi + (long)zl * bLo;
  if (C)    C   += (long)zh * cHi + (long)zl * cLo;
  if (Cbf)  Cbf += (long)zh * cbHi + (long)zl * cbLo;
  if (bias) bias += (long)zh * biasHi + (long)zl * biasLo;

  int m0 = blockIdx.y * 64, n0 = blockIdx.x * 64;
  int tid = threadIdx.x, wave = tid >> 5, lane = tid & 31;
  int wm = wave >> 1, wn = wave & 1;
  int half = lane >> 4, idx = lane & 15;
  int kb = half * 8;
  v8f acc0 = {}, acc1 = {};

  for (int k0 = 0; k0 < K; k0 += 32) {
    __syncthreads();                       // previous tile fully consumed
#if USE_TDM
    if (wave == 0)
      tdm_load_2d((unsigned int)(uintptr_t)&As[0][0],
                  A + (long)m0 * lda + k0, 32, 64,
                  (unsigned int)lda, (unsigned int)(K - k0),
                  (unsigned int)(M - m0), 1);
    else if (wave == 1)
      tdm_load_2d((unsigned int)(uintptr_t)&Bs[0][0],
                  Bm + (long)n0 * ldb + k0, 32, 64,
                  (unsigned int)ldb, (unsigned int)(K - k0),
                  (unsigned int)(N - n0), 1);
    if (wave < 2) __builtin_amdgcn_s_wait_tensorcnt(0);
#else
    for (int i = tid; i < 64 * 32; i += 256) {
      int r = i >> 5, c = i & 31;
      int gm = m0 + r, gk = k0 + c;
      As[r][c] = (gm < M && gk < K) ? A[(long)gm * lda + gk] : (bf16)0.f;
    }
    if (k0 + 32 < K && tid < 64 && (m0 + tid) < M)
      __builtin_prefetch(&A[(long)(m0 + tid) * lda + k0 + 32], 0, 1);
    for (int i = tid; i < 64 * 32; i += 256) {
      int r = i >> 5, c = i & 31;
      int gn = n0 + r, gk = k0 + c;
      Bs[r][c] = (gn < N && gk < K) ? Bm[(long)gn * ldb + gk] : (bf16)0.f;
    }
#endif
    __syncthreads();
    v16bf af, b0f, b1f;
    int ar  = wm * 16 + idx;
    int br0 = wn * 32 + idx;
    int br1 = br0 + 16;
#pragma unroll
    for (int e = 0; e < 8; ++e) {
      af[e]      = As[ar][kb + e];       af[8 + e]  = As[ar][16 + kb + e];
      b0f[e]     = Bs[br0][kb + e];      b0f[8 + e] = Bs[br0][16 + kb + e];
      b1f[e]     = Bs[br1][kb + e];      b1f[8 + e] = Bs[br1][16 + kb + e];
    }
    acc0 = __builtin_amdgcn_wmma_f32_16x16x32_bf16(false, af, false, b0f,
                                                   (short)0, acc0, false, false);
    acc1 = __builtin_amdgcn_wmma_f32_16x16x32_bf16(false, af, false, b1f,
                                                   (short)0, acc1, false, false);
  }
#pragma unroll
  for (int i = 0; i < 8; ++i) {
    int gm = m0 + wm * 16 + half * 8 + i;
    if (gm >= M) continue;
#pragma unroll
    for (int t = 0; t < 2; ++t) {
      int gn = n0 + wn * 32 + idx + t * 16;
      if (gn >= N) continue;
      float v = (t ? acc1[i] : acc0[i]) * alpha + (bias ? bias[gn] : 0.f);
      if (act) v = fmaxf(v, 0.f);
      if (C) C[(long)gm * ldc + gn] = v;
      if (Cbf) {
        if (transCbf) Cbf[(long)gn * ldcb + gm] = (bf16)v;
        else          Cbf[(long)gm * ldcb + gn] = (bf16)v;
      }
    }
  }
}

// ---------------------------------------------------------------------------
// Persistent LSTM direction (1 block/dir). Whh (bf16, 100KB) TDM-preloaded
// into LDS once; h/c live in LDS; recurrent 32x400xK128 matmul via WMMA.
// ---------------------------------------------------------------------------
__device__ inline float sigmoidf_(float x) { return 1.f / (1.f + __expf(-x)); }

__global__ __launch_bounds__(256) void lstm_layer_kernel(
    const float* __restrict__ pre,   // [2][S][B][400] f32
    const bf16*  __restrict__ whhb,  // [2][400][128] bf16 (this layer)
    bf16* __restrict__ xout,         // [S][B][200] bf16
    int S)
{
  __shared__ __align__(16) bf16 wlds[G4][128];
  __shared__ __align__(16) bf16 hbf[32][128];
  __shared__ float gbuf[32][400];
  __shared__ float cst[32][100];
  int dir = blockIdx.x;
  int tid = threadIdx.x, wave = tid >> 5, lane = tid & 31;
  int half = lane >> 4, idx = lane & 15;
  const bf16* wd = whhb + (long)dir * G4 * 128;
#if USE_TDM
  if (wave == 0) {
    tdm_load_2d((unsigned int)(uintptr_t)&wlds[0][0], wd,
                (unsigned int)(G4 * 128), 1, (unsigned int)(G4 * 128),
                (unsigned int)(G4 * 128), 1, 0);
    __builtin_amdgcn_s_wait_tensorcnt(0);
  }
#else
  for (int i = tid; i < G4 * 128; i += 256) wlds[i >> 7][i & 127] = wd[i];
#endif
  for (int i = tid; i < 32 * 128; i += 256) hbf[i >> 7][i & 127] = (bf16)0.f;
  for (int i = tid; i < 32 * RH; i += 256) cst[i / RH][i % RH] = 0.f;
  __syncthreads();
  const float* pd = pre + (long)dir * S * BATCH * G4;
  for (int t = 0; t < S; ++t) {
    int s = dir ? (S - 1 - t) : t;
    const float* ps = pd + (long)s * BATCH * G4;
    for (int tile = wave; tile < 50; tile += 8) {
      int mt = tile & 1, nt = tile >> 1;
      int m0 = mt * 16, n0 = nt * 16;
      v8f acc = {};
      int ar = m0 + idx, br = n0 + idx;
      for (int kk = 0; kk < 4; ++kk) {
        int k0 = kk * 32;
        v16bf af, bfr;
#pragma unroll
        for (int e = 0; e < 8; ++e) {
          af[e]      = hbf[ar][k0 + half * 8 + e];
          af[8 + e]  = hbf[ar][k0 + 16 + half * 8 + e];
          bfr[e]     = wlds[br][k0 + half * 8 + e];
          bfr[8 + e] = wlds[br][k0 + 16 + half * 8 + e];
        }
        acc = __builtin_amdgcn_wmma_f32_16x16x32_bf16(false, af, false, bfr,
                                                      (short)0, acc, false, false);
      }
#pragma unroll
      for (int i = 0; i < 8; ++i)
        gbuf[m0 + half * 8 + i][n0 + idx] =
            ps[(long)(m0 + half * 8 + i) * G4 + (n0 + idx)] + acc[i];
    }
    __syncthreads();
    for (int i = tid; i < 32 * RH; i += 256) {
      int b = i / RH, j = i % RH;
      float ig = sigmoidf_(gbuf[b][j]);
      float fg = sigmoidf_(gbuf[b][RH + j]);
      float gg = tanhf(gbuf[b][2 * RH + j]);
      float og = sigmoidf_(gbuf[b][3 * RH + j]);
      float c  = fg * cst[b][j] + ig * gg;
      float h  = og * tanhf(c);
      cst[b][j] = c;
      hbf[b][j] = (bf16)h;
      xout[((long)s * BATCH + b) * CH + dir * RH + j] = (bf16)h;
    }
    __syncthreads();
  }
}

// ---------------------------------------------------------------------------
// Row softmax in place on f32 pAttn (d_out), dual-writes bf16 mirror.
// ---------------------------------------------------------------------------
__global__ __launch_bounds__(256) void softmax_kernel(float* __restrict__ p,
                                                      bf16* __restrict__ pb, int S)
{
  __shared__ float red[256];
  long base = (long)blockIdx.x * S;
  int tid = threadIdx.x;
  float m = -3.4e38f;
  for (int i = tid; i < S; i += 256) m = fmaxf(m, p[base + i]);
  red[tid] = m; __syncthreads();
  for (int off = 128; off > 0; off >>= 1) {
    if (tid < off) red[tid] = fmaxf(red[tid], red[tid + off]);
    __syncthreads();
  }
  m = red[0]; __syncthreads();
  float sum = 0.f;
  for (int i = tid; i < S; i += 256) {
    float e = __expf(p[base + i] - m);
    p[base + i] = e; sum += e;
  }
  red[tid] = sum; __syncthreads();
  for (int off = 128; off > 0; off >>= 1) {
    if (tid < off) red[tid] += red[tid + off];
    __syncthreads();
  }
  float inv = 1.f / red[0];
  for (int i = tid; i < S; i += 256) {
    float v = p[base + i] * inv;
    p[base + i] = v;
    pb[base + i] = (bf16)v;
  }
}

__global__ __launch_bounds__(256) void sum_s_kernel(const float* __restrict__ o,
                                                    float* __restrict__ osum)
{
  int i = blockIdx.x * 256 + threadIdx.x;
  if (i >= BATCH * RH) return;
  int b = i / RH, j = i % RH;
  float s = 0.f;
  const float* op = o + (long)b * S_TOT * RH + j;
  for (int t = 0; t < S_TOT; ++t) s += op[(long)t * RH];
  osum[i] = s;
}

__global__ __launch_bounds__(256) void head_kernel(
    const float* __restrict__ osum, const float* __restrict__ fc3w,
    const float* __restrict__ fc3b, float* __restrict__ out)
{
  __shared__ float r1[256], r2[256];
  int tid = threadIdx.x;
  float s = 0.f, s2 = 0.f;
  for (int i = tid; i < BATCH * RH; i += 256) {
    float v = osum[i]; s += v; s2 += v * v;
  }
  r1[tid] = s; r2[tid] = s2; __syncthreads();
  for (int off = 128; off > 0; off >>= 1) {
    if (tid < off) { r1[tid] += r1[tid + off]; r2[tid] += r2[tid + off]; }
    __syncthreads();
  }
  float n = (float)(BATCH * RH);
  float mean = r1[0] / n;
  float var = (r2[0] - n * mean * mean) / (n - 1.f);
  float invstd = rsqrtf(var);
  if (tid < BATCH * 2) {
    int b = tid >> 1, j2 = tid & 1;
    float acc = fc3b[j2];
    for (int j = 0; j < RH; ++j)
      acc += (osum[b * RH + j] - mean) * invstd * fc3w[j * 2 + j2];
    out[tid] = acc;
  }
}

// ---------------------------------------------------------------------------
extern "C" void kernel_launch(void* const* d_in, const int* in_sizes, int n_in,
                              void* d_out, int out_size, void* d_ws, size_t ws_size,
                              hipStream_t stream)
{
  const float* input_p  = (const float*)d_in[0];
  const float* input_e  = (const float*)d_in[1];
  const float* conv_e_w = (const float*)d_in[2];
  const float* bn_e_g   = (const float*)d_in[3];
  const float* bn_e_b   = (const float*)d_in[4];
  const float* bn_e_m   = (const float*)d_in[5];
  const float* bn_e_v   = (const float*)d_in[6];
  const float* conv_p_w = (const float*)d_in[7];
  const float* bn_p_g   = (const float*)d_in[8];
  const float* bn_p_b   = (const float*)d_in[9];
  const float* bn_p_m   = (const float*)d_in[10];
  const float* bn_p_v   = (const float*)d_in[11];
  const float* lstm_wih = (const float*)d_in[12];
  const float* lstm_whh = (const float*)d_in[13];
  const float* lstm_bih = (const float*)d_in[14];
  const float* lstm_bhh = (const float*)d_in[15];
  const float* Qw = (const float*)d_in[16];
  const float* Qb = (const float*)d_in[17];
  const float* Kw = (const float*)d_in[18];
  const float* Kb = (const float*)d_in[19];
  const float* Vw = (const float*)d_in[20];
  const float* Vb = (const float*)d_in[21];
  const float* mh_w  = (const float*)d_in[22];
  const float* mh_b  = (const float*)d_in[23];
  const float* fc3_w = (const float*)d_in[24];
  const float* fc3_b = (const float*)d_in[25];
  float* out = (float*)d_out;
  float* ws  = (float*)d_ws;

  const long SBC  = (long)S_TOT * BATCH * CH;        // 5,305,600
  const long SBG  = (long)S_TOT * BATCH * G4;        // 10,611,200
  const long QKV  = (long)BATCH * NH * S_TOT * HD;   // 6,791,168
  const long PAT  = (long)BATCH * S_TOT * NH * S_TOT;
  // float-unit offsets (bf16 regions counted as half)
  long o_x0b  = 0;                       // bf16 SBC
  long o_x1b  = o_x0b + SBC / 2;         // bf16 SBC
  long o_pre  = o_x1b + SBC / 2;         // f32 2*SBG
  long o_wihb = o_pre + 2 * SBG;         // bf16 2*2*400*200
  long o_whhb = o_wihb + (2L * 2 * G4 * CH) / 2;
  long o_qwT  = o_whhb + (2L * 2 * G4 * 128) / 2;
  long o_kwT  = o_qwT + (long)NH * HD * CH / 2;
  long o_vwT  = o_kwT + (long)NH * HD * CH / 2;
  long o_mhwT = o_vwT + (long)NH * HD * CH / 2;
  long o_bias = o_mhwT + (long)RH * NH * HD / 2;
  long o_qb   = o_bias + 2L * 2 * G4;    // bf16 QKV
  long o_kb   = o_qb + QKV / 2;
  long o_vbT  = o_kb + QKV / 2;          // v stored transposed (b,h,32,S)
  long o_attc = o_vbT + QKV / 2;         // bf16 (B,S,256)
  long o_pab  = o_attc + (long)BATCH * S_TOT * NH * HD / 2;  // bf16 p_attn mirror
  long o_ob   = o_pab + PAT / 2;         // f32 (B,S,100)
  long o_osum = o_ob + (long)BATCH * S_TOT * RH;

  bf16* x0b  = (bf16*)(ws + o_x0b);
  bf16* x1b  = (bf16*)(ws + o_x1b);
  bf16* wihb = (bf16*)(ws + o_wihb);
  bf16* whhb = (bf16*)(ws + o_whhb);
  bf16* qwT  = (bf16*)(ws + o_qwT);
  bf16* kwT  = (bf16*)(ws + o_kwT);
  bf16* vwT  = (bf16*)(ws + o_vwT);
  bf16* mhwT = (bf16*)(ws + o_mhwT);
  bf16* qb   = (bf16*)(ws + o_qb);
  bf16* kb   = (bf16*)(ws + o_kb);
  bf16* vbT  = (bf16*)(ws + o_vbT);
  bf16* attc = (bf16*)(ws + o_attc);
  bf16* pab  = (bf16*)(ws + o_pab);

  auto gemm = [&](const bf16* A, const bf16* Bm, const float* bias,
                  float* C, bf16* Cbf, int transCbf,
                  int M, int N, int K, int lda, int ldb, int ldc, int ldcb,
                  float alpha, int act, int batch, int zdiv,
                  long aHi, long aLo, long bHi, long bLo, long cHi, long cLo,
                  long cbHi, long cbLo, long biasHi, long biasLo) {
    dim3 g((N + 63) / 64, (M + 63) / 64, batch);
    gemm_wmma_kernel<<<g, 256, 0, stream>>>(A, Bm, bias, C, Cbf, transCbf,
        M, N, K, lda, ldb, ldc, ldcb, alpha, act, zdiv,
        aHi, aLo, bHi, bLo, cHi, cLo, cbHi, cbLo, biasHi, biasLo);
  };

  // 1) weight prep (cast/transpose to bf16)
  prep_kernel<<<(2 * 2 * G4 * CH + 255) / 256, 256, 0, stream>>>(
      lstm_wih, lstm_whh, lstm_bih, lstm_bhh, Qw, Kw, Vw, mh_w,
      wihb, whhb, qwT, kwT, vwT, mhwT, ws + o_bias);

  // 2) conv branches -> x0 bf16 (S,B,200)
  conv_branch_kernel<<<(498 * BATCH * CH + 255) / 256, 256, 0, stream>>>(
      input_p, conv_e_w, bn_e_g, bn_e_b, bn_e_m, bn_e_v, x0b, 3000, 498, 0);
  conv_branch_kernel<<<(331 * BATCH * CH + 255) / 256, 256, 0, stream>>>(
      input_e, conv_p_w, bn_p_g, bn_p_b, bn_p_m, bn_p_v, x0b, 2000, 331, 498);

  const int MSB = S_TOT * BATCH;  // 26528

  // 3) LSTM layer 0
  gemm(x0b, wihb, ws + o_bias, ws + o_pre, nullptr, 0,
       MSB, G4, CH, CH, CH, G4, 0, 1.f, 0, 2, 1,
       0, 0, (long)G4 * CH, 0, SBG, 0, 0, 0, (long)G4, 0);
  lstm_layer_kernel<<<2, 256, 0, stream>>>(ws + o_pre, whhb, x1b, S_TOT);

  // 4) LSTM layer 1 -> x2 reuses x0b
  gemm(x1b, wihb + 2L * G4 * CH, ws + o_bias + 2 * G4, ws + o_pre, nullptr, 0,
       MSB, G4, CH, CH, CH, G4, 0, 1.f, 0, 2, 1,
       0, 0, (long)G4 * CH, 0, SBG, 0, 0, 0, (long)G4, 0);
  lstm_layer_kernel<<<2, 256, 0, stream>>>(ws + o_pre, whhb + 2L * G4 * 128,
                                           x0b, S_TOT);
  bf16* x2b = x0b;  // (S,B,200) bf16

  // 5) Q/K/V projections: batch z = b*8+h; A rows are s with lda=B*200
  const long qHi = (long)NH * S_TOT * HD, qLo = (long)S_TOT * HD;
  gemm(x2b, qwT, Qb, nullptr, qb, 0, S_TOT, HD, CH, BATCH * CH, CH, 0, HD,
       1.f, 0, BATCH * NH, NH, CH, 0, 0, (long)HD * CH, 0, 0, qHi, qLo, 0, HD);
  gemm(x2b, kwT, Kb, nullptr, kb, 0, S_TOT, HD, CH, BATCH * CH, CH, 0, HD,
       1.f, 0, BATCH * NH, NH, CH, 0, 0, (long)HD * CH, 0, 0, qHi, qLo, 0, HD);
  gemm(x2b, vwT, Vb, nullptr, vbT, /*transC=*/1, S_TOT, HD, CH, BATCH * CH, CH,
       0, S_TOT, 1.f, 0, BATCH * NH, NH, CH, 0, 0, (long)HD * CH, 0, 0,
       qHi, qLo, 0, HD);

  // 6) scores = q @ k^T / sqrt(32) -> f32 pAttn region of d_out
  float* pattn = out + 64;
  const long sHi = (long)NH * S_TOT * S_TOT;
  gemm(qb, kb, nullptr, pattn, nullptr, 0,
       S_TOT, S_TOT, HD, HD, HD, NH * S_TOT, 0,
       0.1767766953f, 0, BATCH * NH, NH,
       qHi, qLo, qHi, qLo, sHi, (long)S_TOT, 0, 0, 0, 0);

  // 7) softmax in place + bf16 mirror
  softmax_kernel<<<BATCH * S_TOT * NH, 256, 0, stream>>>(pattn, pab, S_TOT);

  // 8) attn = relu(p_attn @ v) -> bf16 attn_concat (B,S,256)
  gemm(pab, vbT, nullptr, nullptr, attc, 0,
       S_TOT, HD, S_TOT, NH * S_TOT, S_TOT, 0, NH * HD,
       1.f, 1, BATCH * NH, NH,
       sHi, (long)S_TOT, (long)NH * HD * S_TOT, (long)HD * S_TOT,
       0, 0, (long)S_TOT * NH * HD, (long)HD, 0, 0);

  // 9) o = relu(attn_concat @ mh_w + mh_b) f32 (B*S,100)
  gemm(attc, mhwT, mh_b, ws + o_ob, nullptr, 0,
       MSB, RH, NH * HD, NH * HD, NH * HD, RH, 0,
       1.f, 1, 1, 1, 0, 0, 0, 0, 0, 0, 0, 0, 0, 0);

  // 10) sum over S, normalize, fc3 -> logits d_out[0:64]
  sum_s_kernel<<<(BATCH * RH + 255) / 256, 256, 0, stream>>>(ws + o_ob, ws + o_osum);
  head_kernel<<<1, 256, 0, stream>>>(ws + o_osum, fc3_w, fc3_b, out);
}
